// LaplacianTopKSAE_32461362823170
// MI455X (gfx1250) — compile-verified, matrix operand order checked by reading, not run on hardware
//
#include <hip/hip_runtime.h>
#include <stdint.h>

// ---------------- problem constants ----------------
#define B_ROWS 4096
#define D_DIM  1024
#define F_DIM  16384
#define K_SEL  64

// ---------------- tiling ----------------
#define TILE_M   64                 // rows of x per block (4 WMMA M-tiles)
#define WAVES    8                  // 256 threads, wave32
#define CHUNK_N  (WAVES * 16)       // 128 feature columns per F-iteration
#define KSTEP    32                 // bf16 WMMA K depth
#define FSPLIT   4                  // F-dimension slices (occupancy)
#define F_SLICE  (F_DIM / FSPLIT)   // 4096 features per block

typedef __attribute__((ext_vector_type(16))) __bf16 v16bf;
typedef __attribute__((ext_vector_type(8)))  float  v8f;
typedef __attribute__((ext_vector_type(4)))  unsigned int u32x4;
typedef __attribute__((ext_vector_type(8)))  int i32x8;
typedef __attribute__((ext_vector_type(4)))  int i32x4;

// ---------------- LDS layout (dynamic, 160.5 KB < 320 KB WGP LDS) ----------------
#define XT_BYTES  (TILE_M * D_DIM * 2)        // 131072: x tile as bf16 (TDM target)
#define TV_BYTES  (TILE_M * K_SEL * 4)        // 16384 : top-k signed values
#define TI_BYTES  (TILE_M * K_SEL * 4)        // 16384 : top-k feature indices
#define SMEM_TOTAL (XT_BYTES + TV_BYTES + TI_BYTES + TILE_M * 4 + TILE_M * 4)

__device__ __forceinline__ unsigned short f2bf(float f) {
  __bf16 h = (__bf16)f;
  unsigned short u;
  __builtin_memcpy(&u, &h, 2);
  return u;
}

// ------------------------------------------------------------------
// Kernel 0: f32 -> bf16 pack (one launch for x, one for W_enc).
// HBM-bound: 80 MB read + 40 MB write ~= 5 us at 23.3 TB/s.
// ------------------------------------------------------------------
__global__ void __launch_bounds__(256)
cvt_bf16(const float* __restrict__ src, unsigned short* __restrict__ dst, int n4)
{
  int i = blockIdx.x * 256 + threadIdx.x;
  if (i < n4) {
    float4 v = ((const float4*)src)[i];
    uint2 o;
    o.x = (unsigned)f2bf(v.x) | ((unsigned)f2bf(v.y) << 16);
    o.y = (unsigned)f2bf(v.z) | ((unsigned)f2bf(v.w) << 16);
    ((uint2*)dst)[i] = o;
  }
}

// Per-row top-64 replace-min insertion. Ballot-serialized so only one lane
// of a wave ever holds the LDS spinlock (wave32-safe); cross-wave contention
// resolved by the lock. Pre-filtered by the racy running min (cold path).
__device__ __attribute__((noinline))
void insert_candidate(float v, int row, int fcol, int lane,
                      float* tvals, int* tidx, float* rowMin, int* rowLk)
{
  float av = fabsf(v);
  bool want = av > rowMin[row];
  unsigned long long m = __ballot(want);
  while (m) {
    int l = __ffsll(m) - 1;
    m &= m - 1;
    if (lane == l) {
      while (atomicCAS(&rowLk[row], 0, 1) != 0) {}
      if (av > rowMin[row]) {
        float* rv = tvals + row * K_SEL;
        int*   ri = tidx  + row * K_SEL;
        int ms = 0; float msv = fabsf(rv[0]);
        for (int s = 1; s < K_SEL; ++s) {
          float t = fabsf(rv[s]);
          if (t < msv) { msv = t; ms = s; }
        }
        if (av > msv) {
          rv[ms] = v; ri[ms] = fcol;
          float nm = fabsf(rv[0]);
          for (int s = 1; s < K_SEL; ++s) {
            float t = fabsf(rv[s]);
            if (t < nm) nm = t;
          }
          rowMin[row] = nm;
        }
      }
      __threadfence_block();
      atomicExch(&rowLk[row], 0);
    }
  }
}

// ------------------------------------------------------------------
// Kernel 1: fused encode GEMM (bf16 WMMA) + per-row abs-top-64 over one
// F slice. Block = 64 rows x 4096 features; grid = (64 row-blocks, 4 slices).
// x tile staged into LDS by the Tensor Data Mover (TDM); W_enc streamed as
// bf16 (L2-resident). 8 waves each own a 16-wide N slice per 128-wide chunk;
// each B fragment is reused across 4 M-tile WMMAs.
// ------------------------------------------------------------------
__global__ void __launch_bounds__(256)
sae_encode_topk(const unsigned short* __restrict__ xbf,   // [B, D] bf16
                const unsigned short* __restrict__ wbf,   // [F, D] bf16
                const float* __restrict__ b_enc,
                float* __restrict__ candVals,             // [B][FSPLIT][64]
                int*   __restrict__ candIdx)
{
  extern __shared__ char smem[];
  unsigned short* xT    = (unsigned short*)smem;
  float*          tvals = (float*)(smem + XT_BYTES);
  int*            tidx  = (int*)  (smem + XT_BYTES + TV_BYTES);
  float*          rowMin= (float*)(smem + XT_BYTES + TV_BYTES + TI_BYTES);
  int*            rowLk = (int*)  (smem + XT_BYTES + TV_BYTES + TI_BYTES + TILE_M * 4);

  const int tid  = threadIdx.x;
  const int wave = tid >> 5;
  const int lane = tid & 31;
  const int rowBase = blockIdx.x * TILE_M;
  const int fSlice  = blockIdx.y;
  const int fBase   = fSlice * F_SLICE;

  // ---- TDM: DMA the 64x1024 bf16 x tile into LDS (one issue, wave 0) ----
  if (tid == 0) {
    unsigned long long ga =
        (unsigned long long)(uintptr_t)(xbf + (size_t)rowBase * D_DIM);
    u32x4 g0;
    g0[0] = 0x1u;                                   // count=1, user mode
    g0[1] = 0u;                                     // lds_addr = 0 (xT at LDS base)
    g0[2] = (unsigned)(ga & 0xFFFFFFFFu);           // global_addr[31:0]
    g0[3] = (unsigned)((ga >> 32) & 0x01FFFFFFu)    // global_addr[56:32]
          | (2u << 30);                             // type = 2 ("image")
    i32x8 g1;
    g1[0] = (int)(1u << 16);                        // data_size=1 (2 bytes), wg_mask=0
    g1[1] = (int)((unsigned)D_DIM << 16);           // tensor_dim0[15:0]  (1024 elems)
    g1[2] = (int)((unsigned)B_ROWS << 16);          // dim0 hi=0 | tensor_dim1[15:0]=4096
    g1[3] = (int)((unsigned)D_DIM << 16);           // dim1 hi=0 | tile_dim0 = 1024
    g1[4] = (int)(unsigned)TILE_M;                  // tile_dim1 = 64 | tile_dim2 = 0
    g1[5] = (int)(unsigned)D_DIM;                   // tensor_dim0_stride = 1024
    g1[6] = 0;                                      // stride0 hi | dim1_stride lo
    g1[7] = 0;                                      // dim1_stride hi
    i32x4 gz4 = (i32x4){0, 0, 0, 0};
    i32x8 gz8 = (i32x8){0, 0, 0, 0, 0, 0, 0, 0};
    __builtin_amdgcn_tensor_load_to_lds(g0, g1, gz4, gz4, gz8, 0);
  }

  // ---- init top-k state while the DMA is in flight ----
  for (int i = tid; i < TILE_M * K_SEL; i += 256) { tvals[i] = 0.0f; tidx[i] = -1; }
  for (int i = tid; i < TILE_M; i += 256)         { rowMin[i] = 0.0f; rowLk[i] = 0; }

  if (wave == 0) __builtin_amdgcn_s_wait_tensorcnt(0);
  __syncthreads();

  const int nCol  = lane & 15;                 // column within the 16-wide N tile
  const int mHi   = (lane >> 4) ? 8 : 0;       // C/D layout: lanes 16-31 hold M+8
  const int kHalf = (lane >> 4) << 4;          // 16-element K split across half-waves

  #pragma unroll 1
  for (int f0 = fBase + wave * 16; f0 < fBase + F_SLICE; f0 += CHUNK_N) {
    v8f acc[4];
    #pragma unroll
    for (int a = 0; a < 4; ++a) acc[a] = (v8f){};

    const unsigned short* wB = wbf + (size_t)(f0 + nCol) * D_DIM + kHalf;

    #pragma unroll 1
    for (int k = 0; k < D_DIM; k += KSTEP) {
      // B fragment: 16 bf16 (32 B) straight from L2-resident W_enc-bf16
      union { uint4 q[2]; v16bf v; } bu;
      const uint4* bp = (const uint4*)(wB + k);
      bu.q[0] = bp[0]; bu.q[1] = bp[1];

      const int kOff = k + kHalf;
      #pragma unroll
      for (int a = 0; a < 4; ++a) {
        union { uint4 q[2]; v16bf v; } au;
        const uint4* ap = (const uint4*)&xT[(a * 16 + nCol) * D_DIM + kOff];
        au.q[0] = ap[0]; au.q[1] = ap[1];
        acc[a] = __builtin_amdgcn_wmma_f32_16x16x32_bf16(
            false, au.v, false, bu.v, (short)0, acc[a], false, false);
      }
    }

    // ---- top-k insertion for this 64x16 z tile ----
    const int   fcol = f0 + nCol;
    const float be   = b_enc[fcol];
    #pragma unroll
    for (int a = 0; a < 4; ++a) {
      float tmp[8];
      #pragma unroll
      for (int r = 0; r < 8; ++r) tmp[r] = acc[a][r];
      #pragma unroll
      for (int r = 0; r < 8; ++r) {
        insert_candidate(tmp[r] + be, a * 16 + r + mHi, fcol, lane,
                         tvals, tidx, rowMin, rowLk);
      }
    }
  }

  __syncthreads();
  // ---- dump per-row per-slice top-64 to workspace ----
  for (int i = tid; i < TILE_M * K_SEL; i += 256) {
    int row = i / K_SEL, s = i % K_SEL;
    size_t o = ((size_t)(rowBase + row) * FSPLIT + fSlice) * K_SEL + s;
    candVals[o] = tvals[i];
    candIdx [o] = tidx[i];
  }
}

// ------------------------------------------------------------------
// Kernel 2: exact top-64 merge of the FSPLIT*64 = 256 per-slice candidates.
// One block per row; iterative parallel argmax over LDS.
// ------------------------------------------------------------------
__global__ void __launch_bounds__(256)
sae_merge(const float* __restrict__ candVals, const int* __restrict__ candIdx,
          float* __restrict__ selVals, int* __restrict__ selIdx)
{
  __shared__ float sv[FSPLIT * K_SEL];
  __shared__ int   si[FSPLIT * K_SEL];
  __shared__ float av[FSPLIT * K_SEL];
  __shared__ float red [256];
  __shared__ int   redi[256];

  const int b = blockIdx.x, tid = threadIdx.x;
  sv[tid] = candVals[(size_t)b * FSPLIT * K_SEL + tid];
  si[tid] = candIdx [(size_t)b * FSPLIT * K_SEL + tid];
  av[tid] = fabsf(sv[tid]);
  __syncthreads();

  for (int out = 0; out < K_SEL; ++out) {
    red[tid] = av[tid]; redi[tid] = tid;
    __syncthreads();
    #pragma unroll
    for (int s = 128; s > 0; s >>= 1) {
      if (tid < s && red[tid + s] > red[tid]) {
        red[tid] = red[tid + s]; redi[tid] = redi[tid + s];
      }
      __syncthreads();
    }
    if (tid == 0) {
      int w = redi[0];
      selVals[(size_t)b * K_SEL + out] = sv[w];
      selIdx [(size_t)b * K_SEL + out] = si[w];
      av[w] = -1.0f;
    }
    __syncthreads();
  }
}

// ------------------------------------------------------------------
// Kernel 3: sparse decode. x_hat[b,:] = b_dec + sum_k val_k * W_dec[:,f_k].
// W_dec == W_enc^T, so column f of W_dec is the contiguous f32 row
// W_enc[f,:]: coalesced float4 gathers, W_enc L2-resident, full f32 math.
// ------------------------------------------------------------------
__global__ void __launch_bounds__(256)
sae_decode(const float* __restrict__ selVals,
           const int*   __restrict__ selIdx,
           const float* __restrict__ W_enc,    // [F, D] == W_dec^T (f32)
           const float* __restrict__ b_dec,
           float* __restrict__ out)
{
  __shared__ float sv[K_SEL];
  __shared__ int   si[K_SEL];
  const int b = blockIdx.x, tid = threadIdx.x;
  if (tid < K_SEL) {
    sv[tid] = selVals[(size_t)b * K_SEL + tid];
    si[tid] = selIdx [(size_t)b * K_SEL + tid];
  }
  __syncthreads();

  float4 bd = ((const float4*)b_dec)[tid];          // D/4 = 256 lanes exactly
  float ax = bd.x, ay = bd.y, az = bd.z, aw = bd.w;
  #pragma unroll 4
  for (int k = 0; k < K_SEL; ++k) {
    int f = si[k];
    if (f >= 0) {
      float v = sv[k];
      float4 w = ((const float4*)(W_enc + (size_t)f * D_DIM))[tid];
      ax = fmaf(v, w.x, ax); ay = fmaf(v, w.y, ay);
      az = fmaf(v, w.z, az); aw = fmaf(v, w.w, aw);
    }
  }
  float4 r; r.x = ax; r.y = ay; r.z = az; r.w = aw;
  ((float4*)out)[(size_t)b * (D_DIM / 4) + tid] = r;
}

// ------------------------------------------------------------------
extern "C" void kernel_launch(void* const* d_in, const int* in_sizes, int n_in,
                              void* d_out, int out_size, void* d_ws, size_t ws_size,
                              hipStream_t stream) {
  (void)in_sizes; (void)n_in; (void)out_size; (void)ws_size;
  const float* x     = (const float*)d_in[0];
  const float* W_enc = (const float*)d_in[1];   // [F, D]
  const float* b_enc = (const float*)d_in[2];   // [F]
  // d_in[3] = W_dec [D, F]: unused — equals W_enc^T; decode reads W_enc rows.
  const float* b_dec = (const float*)d_in[4];   // [D]
  // d_in[5] = k (==64), baked into K_SEL.
  float* out = (float*)d_out;

  // ---- workspace carve (50 MB) ----
  char* ws = (char*)d_ws;
  unsigned short* xbf = (unsigned short*)ws;                        // 8 MB
  unsigned short* wbf = (unsigned short*)(ws + (size_t)B_ROWS * D_DIM * 2);
  char* p = ws + (size_t)B_ROWS * D_DIM * 2 + (size_t)F_DIM * D_DIM * 2;
  float* candVals = (float*)p;                 p += (size_t)B_ROWS * FSPLIT * K_SEL * 4;
  int*   candIdx  = (int*)p;                   p += (size_t)B_ROWS * FSPLIT * K_SEL * 4;
  float* selVals  = (float*)p;                 p += (size_t)B_ROWS * K_SEL * 4;
  int*   selIdx   = (int*)p;

  dim3 blk(256);

  // 0) pack x and W_enc to bf16 (HBM-bound, ~5 us)
  int nx4 = B_ROWS * D_DIM / 4;
  int nw4 = F_DIM * D_DIM / 4;
  hipLaunchKernelGGL(cvt_bf16, dim3((nx4 + 255) / 256), blk, 0, stream, x, xbf, nx4);
  hipLaunchKernelGGL(cvt_bf16, dim3((nw4 + 255) / 256), blk, 0, stream, W_enc, wbf, nw4);

  // 1) fused encode + per-slice top-64 (256 blocks)
  dim3 g1(B_ROWS / TILE_M, FSPLIT);
  hipLaunchKernelGGL(sae_encode_topk, g1, blk, SMEM_TOTAL, stream,
                     xbf, wbf, b_enc, candVals, candIdx);

  // 2) exact merge to global top-64
  hipLaunchKernelGGL(sae_merge, dim3(B_ROWS), blk, 0, stream,
                     candVals, candIdx, selVals, selIdx);

  // 3) sparse decode (f32)
  hipLaunchKernelGGL(sae_decode, dim3(B_ROWS), blk, 0, stream,
                     selVals, selIdx, W_enc, b_dec, out);
}
